// NetworkV2_35837207117908
// MI455X (gfx1250) — matrix operand order repaired and checked by reading, not compiled, not run on hardware
//
#include <hip/hip_runtime.h>
#include <hip/hip_bf16.h>

// ---------------------------------------------------------------------------
// Model constants (from reference)
// ---------------------------------------------------------------------------
#define DMODEL  512
#define NHEAD   8
#define DHEAD   64
#define DINNER  512          // NHEAD * DHEAD
#define QKVDIM  1536         // 3 * DINNER
#define SEQLEN  2048
#define BATCH   32
#define DEPTH   6
#define NACT    600

typedef __attribute__((ext_vector_type(16))) __bf16 v16bf;
typedef __attribute__((ext_vector_type(8)))  float  v8f;
typedef __attribute__((ext_vector_type(4)))  int    v4i_;

// CDNA5 async global->LDS staging (ASYNCcnt path), guarded so both host pass
// and any toolchain lacking the builtin still compile (manual copy fallback).
// Probe result (round 2 diagnostic): this toolchain declares the builtin with
// generic `int ext_vector(4)*` params (no explicit address-space quals).
#if defined(__has_builtin)
# if __has_builtin(__builtin_amdgcn_global_load_async_to_lds_b128)
#  define HAVE_ASYNC_LDS 1
# endif
#endif
#ifndef HAVE_ASYNC_LDS
# define HAVE_ASYNC_LDS 0
#endif

static __device__ __forceinline__ void wait_async0() {
#if HAVE_ASYNC_LDS
# if __has_builtin(__builtin_amdgcn_s_wait_asynccnt)
    __builtin_amdgcn_s_wait_asynccnt(0);
# else
    asm volatile("s_wait_asynccnt 0x0" ::: "memory");
# endif
#endif
}

// ---------------------------------------------------------------------------
// bf16 helpers (round-to-nearest-even)
// ---------------------------------------------------------------------------
static __device__ __forceinline__ __bf16 f2bf(float f) {
    union { float f; unsigned u; } a; a.f = f;
    unsigned r = a.u + 0x7FFFu + ((a.u >> 16) & 1u);
    union { unsigned short s; __bf16 b; } o; o.s = (unsigned short)(r >> 16);
    return o.b;
}
static __device__ __forceinline__ float bf2f(__bf16 b) {
    union { unsigned short s; __bf16 b; } i; i.b = b;
    union { unsigned u; float f; } o; o.u = ((unsigned)i.s) << 16;
    return o.f;
}

// ---------------------------------------------------------------------------
// Embedding + positional add -> h (f32) and h (bf16)
// ---------------------------------------------------------------------------
__global__ void __launch_bounds__(256)
embed_kernel(const int* __restrict__ x, const float* __restrict__ emb,
             const float* __restrict__ pos, float* __restrict__ h,
             __bf16* __restrict__ hb)
{
    int row = blockIdx.x;            // b*SEQLEN + n
    int n   = row & (SEQLEN - 1);
    int tok = x[row];
    for (int d = threadIdx.x; d < DMODEL; d += blockDim.x) {
        float v = emb[(size_t)tok * DMODEL + d] + pos[(size_t)n * DMODEL + d];
        h [(size_t)row * DMODEL + d] = v;
        hb[(size_t)row * DMODEL + d] = f2bf(v);
    }
}

// ---------------------------------------------------------------------------
// f32 -> bf16 bulk convert (weights)
// ---------------------------------------------------------------------------
__global__ void __launch_bounds__(256)
cvt_bf16_kernel(const float* __restrict__ src, __bf16* __restrict__ dst, size_t n)
{
    size_t i = (size_t)blockIdx.x * blockDim.x + threadIdx.x;
    size_t stride = (size_t)gridDim.x * blockDim.x;
    for (; i < n; i += stride) dst[i] = f2bf(src[i]);
}

// ---------------------------------------------------------------------------
// Tiled bf16 GEMM, f32 WMMA accumulation, double-buffered LDS with async
// global->LDS staging for the A panel.
//   C[M,N] = A[M,K] * W[K,N] (+bias[N]) (+resid[M,N])
// Block tile: 128(M) x 64(N), K-step 32; 8 waves as 4x2, each wave 2x2 WMMAs.
// ---------------------------------------------------------------------------
#define GTM 128
#define GTN 64
#define GTK 32
#define LPAD 8

template <bool BIAS, bool RESID, bool OUTF, bool OUTB>
__global__ void __launch_bounds__(256)
gemm_bf16_kernel(const __bf16* __restrict__ A, const __bf16* __restrict__ W,
                 const float* __restrict__ bias, const float* __restrict__ resid,
                 float* __restrict__ C, __bf16* __restrict__ Cbf,
                 int M, int Nn, int K)
{
    __shared__ __align__(16) __bf16 sA[2][GTM][GTK + LPAD];   // [m][k]
    __shared__ __align__(16) __bf16 sB[2][GTN][GTK + LPAD];   // transposed [n][k]

    const int tid  = threadIdx.x;
    const int lane = tid & 31;
    const int w    = tid >> 5;
    const int wm   = w >> 1, wn = w & 1;
    const int m0   = blockIdx.y * GTM;
    const int n0   = blockIdx.x * GTN;

    // stage A tile (128x32 bf16 = 512 x b128, 2 per thread)
    auto stageA = [&](int k0, int buf) {
        #pragma unroll
        for (int i = 0; i < 2; ++i) {
            int idx = tid + i * 256;
            int r   = idx >> 2, seg = idx & 3;
            const __bf16* gp = &A[(size_t)(m0 + r) * K + k0 + seg * 8];
            __bf16*       lp = &sA[buf][r][seg * 8];
#if HAVE_ASYNC_LDS
            __builtin_amdgcn_global_load_async_to_lds_b128(
                (v4i_*)(void*)gp, (v4i_*)(void*)lp, 0, 0);
#else
            *(uint4*)lp = *(const uint4*)gp;
            if (i == 0) __builtin_prefetch(gp + GTK, 0, 3);
#endif
        }
    };
    // stage B tile transposed into [n][k] (transform -> cannot be async)
    auto stageB = [&](int k0, int buf) {
        int nn = tid & 63;
        int kk = (tid >> 6) * 8;
        const __bf16* src = &W[(size_t)(k0 + kk) * Nn + n0 + nn];
        #pragma unroll
        for (int j = 0; j < 8; ++j)
            sB[buf][nn][kk + j] = src[(size_t)j * Nn];
    };

    v8f acc[2][2];
    #pragma unroll
    for (int mi = 0; mi < 2; ++mi)
        #pragma unroll
        for (int ni = 0; ni < 2; ++ni)
            #pragma unroll
            for (int r = 0; r < 8; ++r) acc[mi][ni][r] = 0.0f;

    stageA(0, 0);
    stageB(0, 0);

    int buf = 0;
    for (int k0 = 0; k0 < K; k0 += GTK) {
        wait_async0();        // this wave's async tile complete
        __syncthreads();      // all waves' LDS writes visible

        if (k0 + GTK < K) {   // software pipeline: stage next tile now
            stageA(k0 + GTK, buf ^ 1);
            stageB(k0 + GTK, buf ^ 1);
        }

        // ---- fragments (ISA 7.12.2 layouts) + WMMA on current buffer ----
        const int arow0 = wm * 32 + (lane & 15);
        const int acb   = (lane >= 16) ? 8 : 0;
        v16bf afr[2], bfr[2];
        #pragma unroll
        for (int mi = 0; mi < 2; ++mi) {
            int r = arow0 + mi * 16;
            union { uint4 u[2]; v16bf v; } t;
            t.u[0] = *(const uint4*)&sA[buf][r][acb];
            t.u[1] = *(const uint4*)&sA[buf][r][acb + 16];
            afr[mi] = t.v;
        }
        const int bcol0 = wn * 32 + (lane & 15);
        const int bkb   = (lane >= 16) ? 8 : 0;
        #pragma unroll
        for (int ni = 0; ni < 2; ++ni) {
            int c = bcol0 + ni * 16;
            union { uint4 u[2]; v16bf v; } t;
            t.u[0] = *(const uint4*)&sB[buf][c][bkb];
            t.u[1] = *(const uint4*)&sB[buf][c][bkb + 16];
            bfr[ni] = t.v;
        }
        #pragma unroll
        for (int mi = 0; mi < 2; ++mi)
            #pragma unroll
            for (int ni = 0; ni < 2; ++ni)
                acc[mi][ni] = __builtin_amdgcn_wmma_f32_16x16x32_bf16(
                    false, afr[mi], false, bfr[ni], (short)0, acc[mi][ni],
                    false, false);
        buf ^= 1;
    }

    // ---- epilogue: C/D layout VGPR r <-> row r (+8 for lanes>=16) ----
    const int rr = (lane >= 16) ? 8 : 0;
    const int cc = lane & 15;
    #pragma unroll
    for (int mi = 0; mi < 2; ++mi)
        #pragma unroll
        for (int ni = 0; ni < 2; ++ni)
            #pragma unroll
            for (int r = 0; r < 8; ++r) {
                int grow = m0 + wm * 32 + mi * 16 + r + rr;
                int gcol = n0 + wn * 32 + ni * 16 + cc;
                float v = acc[mi][ni][r];
                if (BIAS)  v += bias[gcol];
                size_t off = (size_t)grow * Nn + gcol;
                if (RESID) v += resid[off];
                if (OUTF)  C[off]   = v;
                if (OUTB)  Cbf[off] = f2bf(v);
            }
}

// ---------------------------------------------------------------------------
// Linear attention, phase 1: per (b,h)  kv[64][64] = sum_n k'[n,d] v'[n,e],
// ksum[64] = sum_n k'[n,d].  k' = elu(k)+1 masked, v' = v masked.
// WMMA: A = K^T (d x n), B = V (n x e); reduction over n, 64 steps of 32.
// ---------------------------------------------------------------------------
__global__ void __launch_bounds__(256)
attn_kv_kernel(const __bf16* __restrict__ qkv, const unsigned char* __restrict__ mask,
               float* __restrict__ kv_buf, float* __restrict__ ksum_buf)
{
    const int bh = blockIdx.x, b = bh >> 3, h = bh & 7;
    __shared__ __align__(16) __bf16 sK[DHEAD][GTK + LPAD];  // [d][n]
    __shared__ __align__(16) __bf16 sV[DHEAD][GTK + LPAD];  // [e][n]

    const int tid  = threadIdx.x;
    const int lane = tid & 31;
    const int w    = tid >> 5;

    v8f acc[2];
    #pragma unroll
    for (int t = 0; t < 2; ++t)
        #pragma unroll
        for (int r = 0; r < 8; ++r) acc[t][r] = 0.0f;
    float ksum_acc = 0.0f;

    const int nn = tid >> 3;             // 0..31 chunk-local n
    const int dg = (tid & 7) * 8;        // 8 d's per thread

    for (int n0 = 0; n0 < SEQLEN; n0 += GTK) {
        const int gn = n0 + nn;
        const float mval = mask[(size_t)b * SEQLEN + gn] ? 1.0f : 0.0f;
        const __bf16* krow =
            &qkv[((size_t)b * SEQLEN + gn) * QKVDIM + DINNER + h * DHEAD + dg];
        const __bf16* vrow = krow + DINNER;
        #pragma unroll
        for (int j = 0; j < 8; ++j) {
            float kx = bf2f(krow[j]);
            float vx = bf2f(vrow[j]);
            float kt = (kx > 0.0f ? kx + 1.0f : __expf(kx)) * mval;  // elu+1
            sK[dg + j][nn] = f2bf(kt);
            sV[dg + j][nn] = f2bf(vx * mval);
        }
        __syncthreads();

        if (tid < DHEAD) {
            float s = 0.0f;
            #pragma unroll
            for (int j = 0; j < GTK; ++j) s += bf2f(sK[tid][j]);
            ksum_acc += s;
        }

        const int acb = (lane >= 16) ? 8 : 0;
        const int bkb = (lane >= 16) ? 8 : 0;
        #pragma unroll
        for (int ti = 0; ti < 2; ++ti) {
            int t  = w * 2 + ti;
            int tm = t >> 2, tn = t & 3;
            union { uint4 u[2]; v16bf v; } ta, tb;
            int ar = tm * 16 + (lane & 15);
            ta.u[0] = *(const uint4*)&sK[ar][acb];
            ta.u[1] = *(const uint4*)&sK[ar][acb + 16];
            int bc = tn * 16 + (lane & 15);
            tb.u[0] = *(const uint4*)&sV[bc][bkb];
            tb.u[1] = *(const uint4*)&sV[bc][bkb + 16];
            acc[ti] = __builtin_amdgcn_wmma_f32_16x16x32_bf16(
                false, ta.v, false, tb.v, (short)0, acc[ti], false, false);
        }
        __syncthreads();
    }

    float* kvout = &kv_buf[(size_t)bh * DHEAD * DHEAD];
    const int rr = (lane >= 16) ? 8 : 0;
    const int cc = lane & 15;
    #pragma unroll
    for (int ti = 0; ti < 2; ++ti) {
        int t = w * 2 + ti;
        int tm = t >> 2, tn = t & 3;
        #pragma unroll
        for (int r = 0; r < 8; ++r)
            kvout[(size_t)(tm * 16 + r + rr) * DHEAD + tn * 16 + cc] = acc[ti][r];
    }
    if (tid < DHEAD) ksum_buf[(size_t)bh * DHEAD + tid] = ksum_acc;
}

// ---------------------------------------------------------------------------
// Linear attention, phase 2: out[n,e] = (q'[n,:] @ kv) * z[n],
// z = 1/(q'.ksum + 1e-6). Block per (128-row n-chunk, b); loops heads.
// ---------------------------------------------------------------------------
__global__ void __launch_bounds__(256)
attn_out_kernel(const __bf16* __restrict__ qkv, const float* __restrict__ kv_buf,
                const float* __restrict__ ksum_buf, __bf16* __restrict__ attnb)
{
    const int b  = blockIdx.y;
    const int n0 = blockIdx.x * 128;
    __shared__ __align__(16) __bf16 sQ [128][DHEAD + LPAD];   // [n][d]
    __shared__ __align__(16) __bf16 sKV[DHEAD][DHEAD + LPAD]; // [e][d]
    __shared__ float sKs[DHEAD];
    __shared__ float sZ[128];

    const int tid  = threadIdx.x;
    const int lane = tid & 31;
    const int w    = tid >> 5;

    for (int h = 0; h < NHEAD; ++h) {
        const int bh = b * NHEAD + h;
        {   // stage q' (elu+1)
            int r  = tid >> 1;
            int dg = (tid & 1) * 32;
            const __bf16* qrow =
                &qkv[((size_t)b * SEQLEN + n0 + r) * QKVDIM + h * DHEAD + dg];
            #pragma unroll
            for (int j = 0; j < 32; ++j) {
                float qx = bf2f(qrow[j]);
                sQ[r][dg + j] = f2bf(qx > 0.0f ? qx + 1.0f : __expf(qx));
            }
        }
        {   // stage kv transposed [e][d]
            int e  = tid & 63;
            int dg = (tid >> 6) * 16;
            const float* kvrow = &kv_buf[(size_t)bh * DHEAD * DHEAD];
            #pragma unroll
            for (int j = 0; j < 16; ++j)
                sKV[e][dg + j] = f2bf(kvrow[(size_t)(dg + j) * DHEAD + e]);
        }
        if (tid < DHEAD) sKs[tid] = ksum_buf[(size_t)bh * DHEAD + tid];
        __syncthreads();

        if (tid < 128) {
            float s = 0.0f;
            #pragma unroll
            for (int d = 0; d < DHEAD; ++d) s += bf2f(sQ[tid][d]) * sKs[d];
            sZ[tid] = 1.0f / (s + 1e-6f);
        }
        __syncthreads();

        v8f acc[4];
        #pragma unroll
        for (int tn = 0; tn < 4; ++tn)
            #pragma unroll
            for (int r = 0; r < 8; ++r) acc[tn][r] = 0.0f;

        #pragma unroll
        for (int ks = 0; ks < 2; ++ks) {
            int k0 = ks * 32;
            union { uint4 u[2]; v16bf v; } ta;
            int ar  = w * 16 + (lane & 15);
            int acb = (lane >= 16) ? 8 : 0;
            ta.u[0] = *(const uint4*)&sQ[ar][k0 + acb];
            ta.u[1] = *(const uint4*)&sQ[ar][k0 + acb + 16];
            int bkb = (lane >= 16) ? 8 : 0;
            #pragma unroll
            for (int tn = 0; tn < 4; ++tn) {
                union { uint4 u[2]; v16bf v; } tb;
                int bc = tn * 16 + (lane & 15);
                tb.u[0] = *(const uint4*)&sKV[bc][k0 + bkb];
                tb.u[1] = *(const uint4*)&sKV[bc][k0 + bkb + 16];
                acc[tn] = __builtin_amdgcn_wmma_f32_16x16x32_bf16(
                    false, ta.v, false, tb.v, (short)0, acc[tn], false, false);
            }
        }

        const int rr = (lane >= 16) ? 8 : 0;
        const int cc = lane & 15;
        #pragma unroll
        for (int tn = 0; tn < 4; ++tn)
            #pragma unroll
            for (int r = 0; r < 8; ++r) {
                int row = w * 16 + r + rr;
                int col = tn * 16 + cc;
                float v = acc[tn][r] * sZ[row];
                attnb[((size_t)b * SEQLEN + n0 + row) * DINNER + h * DHEAD + col]
                    = f2bf(v);
            }
        __syncthreads();
    }
}

// ---------------------------------------------------------------------------
// LayerNorm per row
// ---------------------------------------------------------------------------
__global__ void __launch_bounds__(256)
ln_kernel(const float* __restrict__ hin, const float* __restrict__ g,
          const float* __restrict__ be, float* __restrict__ lnout)
{
    const int row = blockIdx.x;
    const int tid = threadIdx.x;
    const int lane = tid & 31, w = tid >> 5;
    __shared__ float s1[8], s2[8];

    float v0 = hin[(size_t)row * DMODEL + tid];
    float v1 = hin[(size_t)row * DMODEL + tid + 256];
    float lsum = v0 + v1, lsq = v0 * v0 + v1 * v1;
    #pragma unroll
    for (int off = 16; off > 0; off >>= 1) {
        lsum += __shfl_down(lsum, off);
        lsq  += __shfl_down(lsq,  off);
    }
    if (lane == 0) { s1[w] = lsum; s2[w] = lsq; }
    __syncthreads();
    if (tid == 0) {
        float a = 0, q = 0;
        #pragma unroll
        for (int i = 0; i < 8; ++i) { a += s1[i]; q += s2[i]; }
        float mean = a / (float)DMODEL;
        float var  = q / (float)DMODEL - mean * mean;
        s1[0] = mean;
        s2[0] = rsqrtf(var + 1e-5f);
    }
    __syncthreads();
    float mean = s1[0], rstd = s2[0];
    lnout[(size_t)row * DMODEL + tid]       = (v0 - mean) * rstd * g[tid]       + be[tid];
    lnout[(size_t)row * DMODEL + tid + 256] = (v1 - mean) * rstd * g[tid + 256] + be[tid + 256];
}

// ---------------------------------------------------------------------------
// Masked mean pooling: block per b, 256 threads x 2 dims each
// ---------------------------------------------------------------------------
__global__ void __launch_bounds__(256)
pool_kernel(const float* __restrict__ lnout, const unsigned char* __restrict__ mask,
            float* __restrict__ pooled)
{
    const int b = blockIdx.x, tid = threadIdx.x;
    const int d0 = tid * 2;
    float a0 = 0, a1 = 0, cnt = 0;
    for (int n = 0; n < SEQLEN; ++n) {
        float m = mask[(size_t)b * SEQLEN + n] ? 1.0f : 0.0f;
        cnt += m;
        const float* r = &lnout[((size_t)b * SEQLEN + n) * DMODEL];
        a0 += m * r[d0];
        a1 += m * r[d0 + 1];
    }
    float len = cnt < 1.0f ? 1.0f : cnt;
    pooled[(size_t)b * DMODEL + d0]     = a0 / len;
    pooled[(size_t)b * DMODEL + d0 + 1] = a1 / len;
}

// ---------------------------------------------------------------------------
// Policy head: pooled @ pol_w + pol_b, softmax over 600. Block per b.
// ---------------------------------------------------------------------------
__global__ void __launch_bounds__(256)
policy_kernel(const float* __restrict__ pooled, const float* __restrict__ w,
              const float* __restrict__ bias, float* __restrict__ out)
{
    const int b = blockIdx.x, tid = threadIdx.x;
    __shared__ float sp[DMODEL];
    __shared__ float slog[NACT];
    __shared__ float red[256];

    for (int d = tid; d < DMODEL; d += 256) sp[d] = pooled[(size_t)b * DMODEL + d];
    __syncthreads();

    for (int c = tid; c < NACT; c += 256) {
        float dot = bias[c];
        for (int d = 0; d < DMODEL; ++d) dot += sp[d] * w[(size_t)d * NACT + c];
        slog[c] = dot;
    }
    __syncthreads();

    float mx = -1e30f;
    for (int c = tid; c < NACT; c += 256) mx = fmaxf(mx, slog[c]);
    red[tid] = mx; __syncthreads();
    for (int s = 128; s > 0; s >>= 1) {
        if (tid < s) red[tid] = fmaxf(red[tid], red[tid + s]);
        __syncthreads();
    }
    mx = red[0]; __syncthreads();

    float part = 0.0f;
    for (int c = tid; c < NACT; c += 256) {
        float e = __expf(slog[c] - mx);
        slog[c] = e;
        part += e;
    }
    red[tid] = part; __syncthreads();
    for (int s = 128; s > 0; s >>= 1) {
        if (tid < s) red[tid] += red[tid + s];
        __syncthreads();
    }
    float inv = 1.0f / red[0];
    for (int c = tid; c < NACT; c += 256)
        out[(size_t)b * NACT + c] = slog[c] * inv;
}

// ---------------------------------------------------------------------------
// Value head: tanh(relu(pooled@v1_w+v1_b) @ v2_w + v2_b). Block per b, 128 thr.
// ---------------------------------------------------------------------------
__global__ void __launch_bounds__(128)
value_kernel(const float* __restrict__ pooled, const float* __restrict__ w1,
             const float* __restrict__ b1, const float* __restrict__ w2,
             const float* __restrict__ b2, float* __restrict__ out)
{
    const int b = blockIdx.x, tid = threadIdx.x;
    __shared__ float sp[DMODEL];
    __shared__ float red[128];
    for (int d = tid; d < DMODEL; d += 128) sp[d] = pooled[(size_t)b * DMODEL + d];
    __syncthreads();
    float dot = b1[tid];
    for (int d = 0; d < DMODEL; ++d) dot += sp[d] * w1[(size_t)d * 128 + tid];
    dot = fmaxf(dot, 0.0f);
    red[tid] = dot * w2[tid];
    __syncthreads();
    for (int s = 64; s > 0; s >>= 1) {
        if (tid < s) red[tid] += red[tid + s];
        __syncthreads();
    }
    if (tid == 0) out[b] = tanhf(red[0] + b2[0]);
}

// ---------------------------------------------------------------------------
// Host launcher
// ---------------------------------------------------------------------------
extern "C" void kernel_launch(void* const* d_in, const int* in_sizes, int n_in,
                              void* d_out, int out_size, void* d_ws, size_t ws_size,
                              hipStream_t stream) {
    (void)in_sizes; (void)n_in; (void)out_size; (void)ws_size;

    const int*           x     = (const int*)d_in[0];
    const unsigned char* mask  = (const unsigned char*)d_in[1];   // jnp bool (1B)
    const float*         emb   = (const float*)d_in[2];
    const float*         pos   = (const float*)d_in[3];
    const float*         qkv_w = (const float*)d_in[4];
    const float*         out_w = (const float*)d_in[5];
    const float*         out_b = (const float*)d_in[6];
    const float*         ln_g  = (const float*)d_in[7];
    const float*         ln_b  = (const float*)d_in[8];
    const float*         pol_w = (const float*)d_in[9];
    const float*         pol_b = (const float*)d_in[10];
    const float*         v1_w  = (const float*)d_in[11];
    const float*         v1_b  = (const float*)d_in[12];
    const float*         v2_w  = (const float*)d_in[13];
    const float*         v2_b  = (const float*)d_in[14];

    // workspace bump allocation (~480 MB total)
    char* p = (char*)d_ws;
    auto alloc = [&](size_t bytes) -> void* {
        void* r = (void*)p;
        p += (bytes + 255) & ~(size_t)255;
        return r;
    };
    const size_t ROWS = (size_t)BATCH * SEQLEN;
    float*  h      = (float*) alloc(ROWS * DMODEL * 4);
    __bf16* hb     = (__bf16*)alloc(ROWS * DMODEL * 2);
    __bf16* qkvb   = (__bf16*)alloc(ROWS * QKVDIM * 2);
    __bf16* attnb  = (__bf16*)alloc(ROWS * DINNER * 2);
    float*  kv_buf = (float*) alloc((size_t)BATCH * NHEAD * DHEAD * DHEAD * 4);
    float*  ksum   = (float*) alloc((size_t)BATCH * NHEAD * DHEAD * 4);
    __bf16* qkvw_b = (__bf16*)alloc((size_t)DEPTH * DMODEL * QKVDIM * 2);
    __bf16* outw_b = (__bf16*)alloc((size_t)DEPTH * DINNER * DMODEL * 2);
    float*  pooled = (float*) alloc((size_t)BATCH * DMODEL * 4);
    float*  lnout  = (float*)qkvb;   // reuse qkv buffer (layers done by then)

    // weight precision conversion
    cvt_bf16_kernel<<<4096, 256, 0, stream>>>(qkv_w, qkvw_b,
                                              (size_t)DEPTH * DMODEL * QKVDIM);
    cvt_bf16_kernel<<<2048, 256, 0, stream>>>(out_w, outw_b,
                                              (size_t)DEPTH * DINNER * DMODEL);

    // embedding + positional
    embed_kernel<<<(int)ROWS, 256, 0, stream>>>(x, emb, pos, h, hb);

    // transformer layers
    for (int l = 0; l < DEPTH; ++l) {
        // qkv = h @ W_qkv  (bf16 out only)
        gemm_bf16_kernel<false, false, false, true>
            <<<dim3(QKVDIM / GTN, (int)(ROWS / GTM)), 256, 0, stream>>>(
            hb, qkvw_b + (size_t)l * DMODEL * QKVDIM,
            nullptr, nullptr, nullptr, qkvb,
            (int)ROWS, QKVDIM, DMODEL);
        // kv = K'^T V', ksum
        attn_kv_kernel<<<BATCH * NHEAD, 256, 0, stream>>>(qkvb, mask, kv_buf, ksum);
        // out = (q' @ kv) * z
        attn_out_kernel<<<dim3(SEQLEN / 128, BATCH), 256, 0, stream>>>(
            qkvb, kv_buf, ksum, attnb);
        // h = h + attn @ W_out + b  (f32 + bf16 mirror)
        gemm_bf16_kernel<true, true, true, true>
            <<<dim3(DMODEL / GTN, (int)(ROWS / GTM)), 256, 0, stream>>>(
            attnb, outw_b + (size_t)l * DINNER * DMODEL,
            out_b + (size_t)l * DMODEL, h, h, hb,
            (int)ROWS, DMODEL, DINNER);
    }

    // LN -> masked mean pool -> heads
    ln_kernel<<<(int)ROWS, 256, 0, stream>>>(h, ln_g, ln_b, lnout);
    pool_kernel<<<BATCH, 256, 0, stream>>>(lnout, mask, pooled);

    float* pol_out = (float*)d_out;
    float* val_out = pol_out + (size_t)BATCH * NACT;
    policy_kernel<<<BATCH, 256, 0, stream>>>(pooled, pol_w, pol_b, pol_out);
    value_kernel<<<BATCH, 128, 0, stream>>>(pooled, v1_w, v1_b, v2_w, v2_b, val_out);
}
